// STCNModel_40827959116552
// MI455X (gfx1250) — matrix operand order repaired and checked by reading, not compiled, not run on hardware
//
#include <hip/hip_runtime.h>

// Problem constants (STCN read_memory)
#define B_   4
#define CK_  64
#define THW_ 12960   // T*H*W = 8*30*54
#define HW_  1620    // H*W  = 30*54
#define N_   8
#define CV_  512

typedef __attribute__((ext_vector_type(16))) __bf16 v16bf;
typedef __attribute__((ext_vector_type(8)))  float  v8f;
typedef __attribute__((ext_vector_type(4)))  float  v4f;
typedef __attribute__((ext_vector_type(4)))  int    v4i;

union FragU { unsigned short s[16]; v4i i[2]; v16bf v; };   // 32B: one bf16 WMMA fragment
union PackU { unsigned short s[8];  v4i v; };               // 16B: 8 packed bf16

// f32 -> bf16 via compiler conversion (lets backend pick native cvt ops)
__device__ __forceinline__ unsigned short f2bf(float f) {
  __bf16 h = (__bf16)f;
  unsigned short s;
  __builtin_memcpy(&s, &h, 2);
  return s;
}

__device__ __forceinline__ v8f wmma_bf16(v16bf a, v16bf b, v8f c) {
  // 8 args: (neg_a, A, neg_b, B, c_mod, C, reuse_a, reuse_b)
  return __builtin_amdgcn_wmma_f32_16x16x32_bf16(false, a, false, b, (short)0, c,
                                                 false, false);
}

// ---------------------------------------------------------------------------
// Prep A: mkT[b][t][c] = bf16(mk[b][c][t]);  asq[b][t] = sum_c mk^2  (fused)
// ---------------------------------------------------------------------------
__global__ __launch_bounds__(256) void stcn_prep_mk(
    const float* __restrict__ mk, float* __restrict__ asq,
    unsigned short* __restrict__ mkT) {
  int idx = blockIdx.x * 256 + threadIdx.x;
  if (idx >= B_ * THW_) return;
  int b = idx / THW_;
  int t = idx - b * THW_;
  const float* p = mk + (size_t)b * CK_ * THW_ + t;
  unsigned short* o = mkT + (size_t)idx * CK_;
  float s = 0.f;
  for (int c0 = 0; c0 < CK_; c0 += 8) {
    PackU pk;
#pragma unroll
    for (int j = 0; j < 8; ++j) {
      float v = p[(size_t)(c0 + j) * THW_];
      s = fmaf(v, v, s);
      pk.s[j] = f2bf(v);
    }
    *(v4i*)(o + c0) = pk.v;
  }
  asq[idx] = s;
}

// ---------------------------------------------------------------------------
// Prep B: qkT[b][q][c] = bf16(qk[b][c][q])
// ---------------------------------------------------------------------------
__global__ __launch_bounds__(256) void stcn_prep_qk(
    const float* __restrict__ qk, unsigned short* __restrict__ qkT) {
  int idx = blockIdx.x * 256 + threadIdx.x;
  if (idx >= B_ * HW_) return;
  int b = idx / HW_;
  int q = idx - b * HW_;
  const float* p = qk + (size_t)b * CK_ * HW_ + q;
  unsigned short* o = qkT + (size_t)idx * CK_;
  for (int c0 = 0; c0 < CK_; c0 += 8) {
    PackU pk;
#pragma unroll
    for (int j = 0; j < 8; ++j) pk.s[j] = f2bf(p[(size_t)(c0 + j) * HW_]);
    *(v4i*)(o + c0) = pk.v;
  }
}

// ---------------------------------------------------------------------------
// Fragment loaders from pre-transposed bf16 (c contiguous, 64 per row).
// A layout (16x32, K=c): lanes 0-15 hold K {0..7,16..23}; hi lanes {8..15,24..31}
// B layout (32x16, K=c): lanes 0-15 hold K {0..7,16..23}; hi lanes {8..15,24..31}
// ---------------------------------------------------------------------------
__device__ __forceinline__ FragU load_afrag(const unsigned short* row, int frag,
                                            bool hi) {
  FragU a;
  const unsigned short* p = row + frag * 32 + (hi ? 8 : 0);
  a.i[0] = *(const v4i*)(p);
  a.i[1] = *(const v4i*)(p + 16);
  return a;
}
__device__ __forceinline__ FragU load_bfrag(const unsigned short* row, int frag,
                                            bool hi) {
  FragU b;
  const unsigned short* p = row + frag * 32 + (hi ? 8 : 0);
  b.i[0] = *(const v4i*)(p);
  b.i[1] = *(const v4i*)(p + 16);
  return b;
}

// ---------------------------------------------------------------------------
// Kernel 1: online-softmax stats per (b, q):  m = max_t S, l = sum_t exp(S-m)
// S[t,q] = (2*mk^T q - a_sq[t]) / 8.  Tile: 32 t x 64 q, 1 subtile/wave.
// ---------------------------------------------------------------------------
__global__ __launch_bounds__(256) void stcn_pass1(
    const unsigned short* __restrict__ mkT, const unsigned short* __restrict__ qkT,
    const float* __restrict__ asq, float* __restrict__ mArr,
    float* __restrict__ lArr) {
  const int b    = blockIdx.y;
  const int q0   = blockIdx.x * 64;
  const int tid  = threadIdx.x;
  const int lane = tid & 31;
  const int w    = tid >> 5;
  const int tb   = w >> 2;   // 0..1 : t sub-block
  const int qb   = w & 3;    // 0..3 : q sub-block
  const int l15  = lane & 15;
  const bool hi  = lane >= 16;

  const unsigned short* mkTb = mkT + (size_t)b * THW_ * CK_;
  const unsigned short* qkTb = qkT + (size_t)b * HW_ * CK_;
  const float* asb = asq + (size_t)b * THW_;

  int qcol = q0 + qb * 16 + l15;
  if (qcol >= HW_) qcol = HW_ - 1;  // clamp: padded columns are discarded

  const FragU bq0 = load_bfrag(qkTb + (size_t)qcol * CK_, 0, hi);
  const FragU bq1 = load_bfrag(qkTb + (size_t)qcol * CK_, 1, hi);

  float m_run = -1e30f, l_run = 0.f;

  for (int t0 = 0; t0 < THW_; t0 += 32) {  // 405 exact iterations
    const int trow = t0 + tb * 16 + l15;
    const unsigned short* arow = mkTb + (size_t)trow * CK_;
    FragU a0 = load_afrag(arow, 0, hi);
    FragU a1 = load_afrag(arow, 1, hi);

    v8f ab = {};
    ab = wmma_bf16(a0.v, bq0.v, ab);
    ab = wmma_bf16(a1.v, bq1.v, ab);

    const int ta = t0 + tb * 16 + (hi ? 8 : 0);
    v4f s0 = *(const v4f*)(asb + ta);
    v4f s1 = *(const v4f*)(asb + ta + 4);

    float sv[8];
#pragma unroll
    for (int r = 0; r < 4; ++r) {
      sv[r]     = (2.f * ab[r]     - s0[r]) * 0.125f;
      sv[r + 4] = (2.f * ab[r + 4] - s1[r]) * 0.125f;
    }
    float mt = sv[0];
#pragma unroll
    for (int r = 1; r < 8; ++r) mt = fmaxf(mt, sv[r]);
    float lt = 0.f;
#pragma unroll
    for (int r = 0; r < 8; ++r) lt += __expf(sv[r] - mt);

    float mn = fmaxf(m_run, mt);
    l_run = l_run * __expf(m_run - mn) + lt * __expf(mt - mn);
    m_run = mn;
  }

  // Combine rows 0..7 with rows 8..15 (lane L <-> lane L+16), wave32 shuffle
  float m2 = __shfl_xor(m_run, 16, 32);
  float l2 = __shfl_xor(l_run, 16, 32);
  float mn = fmaxf(m_run, m2);
  float lc = l_run * __expf(m_run - mn) + l2 * __expf(m2 - mn);

  __shared__ float rm[8][16];
  __shared__ float rl[8][16];
  if (lane < 16) { rm[w][l15] = mn; rl[w][l15] = lc; }
  __syncthreads();
  if (tid < 64) {  // merge the two t-halves (waves qb and qb+4)
    int qb2 = tid >> 4, j = tid & 15;
    float ma = rm[qb2][j], mb2 = rm[qb2 + 4][j];
    float la = rl[qb2][j], lb2 = rl[qb2 + 4][j];
    float mm = fmaxf(ma, mb2);
    float ll = la * __expf(ma - mm) + lb2 * __expf(mb2 - mm);
    int q = q0 + qb2 * 16 + j;
    if (q < HW_) {
      mArr[b * HW_ + q] = mm;
      lArr[b * HW_ + q] = ll;
    }
  }
}

// ---------------------------------------------------------------------------
// Kernel 2: fused value GEMM with on-the-fly softmax.
// out[n,c,q] = (sum_t mv[n,c,t] * exp(S[b,t,q] - m[b,q])) / l[b,q]
// Workgroup tile: 64 CV x 64 q, K loop over t in chunks of 32.
// ---------------------------------------------------------------------------
__global__ __launch_bounds__(256) void stcn_pass2(
    const unsigned short* __restrict__ mkT, const float* __restrict__ mv,
    const unsigned short* __restrict__ qkT, const int* __restrict__ bmap,
    const float* __restrict__ asq, const float* __restrict__ mArr,
    const float* __restrict__ lArr, float* __restrict__ out) {
  // Stride-padded LDS tiles: 40 bf16 (80B) rows -> conflict-free b128 access
  __shared__ __align__(16) unsigned short pT[64 * 40];   // P[q][t]  (bf16)
  __shared__ __align__(16) unsigned short moT[64 * 40];  // V[c][t]  (bf16)

  const int n    = blockIdx.z;
  const int b    = bmap[n];
  const int q0   = blockIdx.x * 64;
  const int cb0  = blockIdx.y * 64;
  const int tid  = threadIdx.x;
  const int lane = tid & 31;
  const int w    = tid >> 5;
  const int tb   = w >> 2;
  const int qb   = w & 3;
  const int l15  = lane & 15;
  const bool hi  = lane >= 16;

  const unsigned short* mkTb = mkT + (size_t)b * THW_ * CK_;
  const unsigned short* qkTb = qkT + (size_t)b * HW_ * CK_;
  const float* asb = asq + (size_t)b * THW_;
  const float* mvn = mv + (size_t)n * CV_ * THW_;

  const int qcol = q0 + qb * 16 + l15;
  const int qc   = qcol < HW_ ? qcol : HW_ - 1;
  const float mval = mArr[b * HW_ + qc];
  const float linv = 1.f / lArr[b * HW_ + qc];

  const FragU bq0 = load_bfrag(qkTb + (size_t)qc * CK_, 0, hi);
  const FragU bq1 = load_bfrag(qkTb + (size_t)qc * CK_, 1, hi);

  v8f acc0 = {}, acc1 = {};

  // Cooperative mem_values staging map: 4 lanes per channel row, 8 t each
  const int mc = tid >> 2;        // 0..63  (CV row in tile)
  const int mt = (tid & 3) * 8;   // 0,8,16,24
  const float* mvrow = mvn + (size_t)(cb0 + mc) * THW_ + mt;

  for (int t0 = 0; t0 < THW_; t0 += 32) {  // 405 exact iterations
    // ---- Phase A: S subtile -> P = exp(S - m) packed bf16 into LDS ----
    const int trow = t0 + tb * 16 + l15;
    const unsigned short* arow = mkTb + (size_t)trow * CK_;
    FragU a0 = load_afrag(arow, 0, hi);
    FragU a1 = load_afrag(arow, 1, hi);

    v8f ab = {};
    ab = wmma_bf16(a0.v, bq0.v, ab);
    ab = wmma_bf16(a1.v, bq1.v, ab);

    const int ta = t0 + tb * 16 + (hi ? 8 : 0);
    v4f s0 = *(const v4f*)(asb + ta);
    v4f s1 = *(const v4f*)(asb + ta + 4);

    PackU pk;
#pragma unroll
    for (int r = 0; r < 4; ++r) {
      pk.s[r]     = f2bf(__expf((2.f * ab[r]     - s0[r]) * 0.125f - mval));
      pk.s[r + 4] = f2bf(__expf((2.f * ab[r + 4] - s1[r]) * 0.125f - mval));
    }
    // Lane holds 8 consecutive t of one q row -> single b128 store
    *(v4i*)&pT[(qb * 16 + l15) * 40 + tb * 16 + (hi ? 8 : 0)] = pk.v;

    // ---- Cooperative mem_values tile (f32 -> bf16) into LDS ----
    v4f va = *(const v4f*)(mvrow + t0);
    v4f vb = *(const v4f*)(mvrow + t0 + 4);
    PackU mpk;
#pragma unroll
    for (int r = 0; r < 4; ++r) {
      mpk.s[r]     = f2bf(va[r]);
      mpk.s[r + 4] = f2bf(vb[r]);
    }
    *(v4i*)&moT[mc * 40 + mt] = mpk.v;

    if (t0 + 32 < THW_) __builtin_prefetch(mvrow + t0 + 32, 0, 1);

    __syncthreads();

    // ---- Phase B: out_tile += V_tile (A) x P_tile (B) ----
    // B fragment (K = t, 32): lanes 0-15 t {0..7,16..23}, lanes 16-31 t {8..15,24..31}
    FragU ubp;
    const int pbase = (qb * 16 + l15) * 40 + (hi ? 8 : 0);
    ubp.i[0] = *(const v4i*)&pT[pbase];
    ubp.i[1] = *(const v4i*)&pT[pbase + 16];

    // A fragments (M = c, K = t): lanes 0-15 t {0..7,16..23}, hi {8..15,24..31}
    FragU ua;
    const int c0r = (tb * 2) * 16 + l15;
    ua.i[0] = *(const v4i*)&moT[c0r * 40 + (hi ? 8 : 0)];
    ua.i[1] = *(const v4i*)&moT[c0r * 40 + (hi ? 8 : 0) + 16];
    acc0 = wmma_bf16(ua.v, ubp.v, acc0);

    const int c1r = (tb * 2 + 1) * 16 + l15;
    ua.i[0] = *(const v4i*)&moT[c1r * 40 + (hi ? 8 : 0)];
    ua.i[1] = *(const v4i*)&moT[c1r * 40 + (hi ? 8 : 0) + 16];
    acc1 = wmma_bf16(ua.v, ubp.v, acc1);

    __syncthreads();
  }

  // ---- Epilogue: divide by softmax denominator, masked store ----
  if (qcol < HW_) {
#pragma unroll
    for (int r = 0; r < 8; ++r) {
      int c = cb0 + (tb * 2) * 16 + (hi ? 8 : 0) + r;
      out[((size_t)n * CV_ + c) * HW_ + qcol] = acc0[r] * linv;
    }
#pragma unroll
    for (int r = 0; r < 8; ++r) {
      int c = cb0 + (tb * 2 + 1) * 16 + (hi ? 8 : 0) + r;
      out[((size_t)n * CV_ + c) * HW_ + qcol] = acc1[r] * linv;
    }
  }
}

// ---------------------------------------------------------------------------
extern "C" void kernel_launch(void* const* d_in, const int* in_sizes, int n_in,
                              void* d_out, int out_size, void* d_ws,
                              size_t ws_size, hipStream_t stream) {
  (void)in_sizes; (void)n_in; (void)out_size; (void)ws_size;
  const float* mk  = (const float*)d_in[0];  // [B,CK,T,H,W]
  const float* mv  = (const float*)d_in[1];  // [N,CV,T,H,W]
  const float* qkp = (const float*)d_in[2];  // [B,CK,H,W]
  const int*   bm  = (const int*)d_in[3];    // [N]
  float* out = (float*)d_out;                // [N,CV,H,W]

  // Workspace: asq[B*THW] f32 | m[B*HW] f32 | l[B*HW] f32 |
  //            mkT[B*THW*CK] bf16 | qkT[B*HW*CK] bf16   (~7.8 MB total)
  float* asq  = (float*)d_ws;
  float* mArr = asq + B_ * THW_;
  float* lArr = mArr + B_ * HW_;
  unsigned short* mkT = (unsigned short*)(lArr + B_ * HW_);
  unsigned short* qkT = mkT + (size_t)B_ * THW_ * CK_;

  stcn_prep_mk<<<dim3((B_ * THW_ + 255) / 256), dim3(256), 0, stream>>>(mk, asq,
                                                                        mkT);
  stcn_prep_qk<<<dim3((B_ * HW_ + 255) / 256), dim3(256), 0, stream>>>(qkp, qkT);
  stcn_pass1<<<dim3((HW_ + 63) / 64, B_), dim3(256), 0, stream>>>(mkT, qkT, asq,
                                                                  mArr, lArr);
  stcn_pass2<<<dim3((HW_ + 63) / 64, CV_ / 64, N_), dim3(256), 0, stream>>>(
      mkT, mv, qkT, bm, asq, mArr, lArr, out);
}
